// Fp6LlmLinear_38397007626929
// MI455X (gfx1250) — compile-verified, matrix operand order checked by reading, not run on hardware
//
#include <hip/hip_runtime.h>
#include <stdint.h>

// FP6-LLM linear: out = x @ (w * scales[:,None]).T + bias   (fp32 accum, fp16 out)
// M=8192 K=4096 N=11008 -> compute-bound fp16 WMMA GEMM for gfx1250 (wave32).

typedef _Float16 half_t;
typedef _Float16 v16h __attribute__((ext_vector_type(16)));
typedef _Float16 v8h  __attribute__((ext_vector_type(8)));
typedef float    v8f  __attribute__((ext_vector_type(8)));

#define BM 128
#define BN 128
#define BK 64
#define PAD 8
#define LDK (BK + PAD)     // 72 halves per LDS row -> 144B stride, bank-conflict free
#define THREADS 256        // 8 waves (wave32)

// byte stride between double-buffer stages (same for shA and shB blocks)
#define STAGE_BYTES (BM * LDK * 2)   // 18432 = 0x4800

// ---- CDNA5 async global->LDS copy (ASYNCcnt tracked), GVS addressing ----
__device__ __forceinline__ void async_load_b128(uint32_t lds_off,
                                                uint32_t goff_bytes,
                                                const void* sbase) {
  asm volatile("global_load_async_to_lds_b128 %0, %1, %2"
               :: "v"(lds_off), "v"(goff_bytes), "s"(sbase)
               : "memory");
}

template <int N>
__device__ __forceinline__ void wait_asynccnt() {
  asm volatile("s_wait_asynccnt %0" :: "n"(N) : "memory");
}

// Build a 16x32 fp16 WMMA fragment for this lane from two aligned 16B LDS chunks.
__device__ __forceinline__ v16h frag16(const half_t* p0, const half_t* p1) {
  union { v16h v; v8h h[2]; } u;
  u.h[0] = *(const v8h*)p0;
  u.h[1] = *(const v8h*)p1;
  return u.v;
}

__global__ __launch_bounds__(THREADS)
void fp6llm_gemm(const half_t* __restrict__ X,   // [M,K]
                 const half_t* __restrict__ W,   // [N,K] (dequantized fp6 values)
                 const half_t* __restrict__ S,   // [N]
                 const half_t* __restrict__ Bb,  // [N]
                 half_t* __restrict__ O,         // [M,N]
                 int M, int N, int K) {
  __shared__ half_t shA[2][BM][LDK];
  __shared__ half_t shB[2][BN][LDK];

  const int tid  = threadIdx.x;
  const int lane = tid & 31;
  const int wave = tid >> 5;
  const int waveM = wave & 3;   // 4 waves along M (4*32 = 128)
  const int waveN = wave >> 2;  // 2 waves along N (2*64 = 128)

  const int blockN = blockIdx.x * BN;
  const int blockM = blockIdx.y * BM;

  // ---- per-thread async-copy plan: 4 b128 chunks per matrix per stage ----
  // tile = 128 rows x 64 halves = 1024 chunks of 8 halves (16B)
  // LDS offsets stored for buffer 0 only; buffer 1 = +STAGE_BYTES (uniform add,
  // avoids m0/v_movrels indexing on the asm operands).
  uint32_t gA[4], gB[4], lA0[4], lB0[4];
#pragma unroll
  for (int i = 0; i < 4; ++i) {
    int c   = tid + i * THREADS;
    int row = c >> 3;
    int kc  = (c & 7) * 8;
    int ra = blockM + row; if (ra > M - 1) ra = M - 1;
    int rb = blockN + row; if (rb > N - 1) rb = N - 1;
    gA[i]  = (uint32_t)(((uint64_t)ra * (uint32_t)K + (uint32_t)kc) * 2u);
    gB[i]  = (uint32_t)(((uint64_t)rb * (uint32_t)K + (uint32_t)kc) * 2u);
    lA0[i] = (uint32_t)(uintptr_t)&shA[0][row][kc];
    lB0[i] = (uint32_t)(uintptr_t)&shB[0][row][kc];
  }

  auto load_stage = [&](int buf, int kt) {
    const uint32_t kadd      = (uint32_t)kt * (BK * 2u);           // bytes
    const uint32_t stage_add = (uint32_t)buf * (uint32_t)STAGE_BYTES;  // uniform
#pragma unroll
    for (int i = 0; i < 4; ++i)
      async_load_b128(lA0[i] + stage_add, gA[i] + kadd, X);
#pragma unroll
    for (int i = 0; i < 4; ++i)
      async_load_b128(lB0[i] + stage_add, gB[i] + kadd, W);
  };

  v8f acc[2][4];
#pragma unroll
  for (int mi = 0; mi < 2; ++mi)
#pragma unroll
    for (int ni = 0; ni < 4; ++ni)
      acc[mi][ni] = (v8f){0.f, 0.f, 0.f, 0.f, 0.f, 0.f, 0.f, 0.f};

  const int KT = K / BK;          // K=4096 -> 64 stages
  load_stage(0, 0);               // 8 async instrs per wave per stage
  if (KT > 1) load_stage(1, 1);

  const int r    = lane & 15;
  const int sel8 = (lane >> 4) * 8;        // lanes 0-15 -> 0, 16-31 -> 8
  const int arow = waveM * 32 + r;
  const int brow = waveN * 64 + r;

  for (int kt = 0; kt < KT; ++kt) {
    const int buf = kt & 1;
    if (kt + 1 < KT) wait_asynccnt<8>();   // only next stage still in flight
    else             wait_asynccnt<0>();
    __syncthreads();

#pragma unroll
    for (int kk = 0; kk < BK; kk += 32) {
      v16h a[2], b[4];
#pragma unroll
      for (int mi = 0; mi < 2; ++mi) {
        const half_t* base = &shA[buf][arow + mi * 16][0];
        a[mi] = frag16(base + kk + sel8, base + kk + 16 + sel8);
      }
#pragma unroll
      for (int ni = 0; ni < 4; ++ni) {
        const half_t* base = &shB[buf][brow + ni * 16][0];
        b[ni] = frag16(base + kk + sel8, base + kk + 16 + sel8);
      }
#pragma unroll
      for (int mi = 0; mi < 2; ++mi)
#pragma unroll
        for (int ni = 0; ni < 4; ++ni)
          acc[mi][ni] = __builtin_amdgcn_wmma_f32_16x16x32_f16(
              false, a[mi], false, b[ni], (short)0, acc[mi][ni], false, false);
    }

    __syncthreads();                       // everyone done reading buf
    if (kt + 2 < KT) load_stage(buf, kt + 2);
  }

  // ---- epilogue: out = fp16(acc * scale[n]) + bias[n] ----
  // C/D layout: VGPR j -> lanes 0-15: M=j, N=lane ; lanes 16-31: M=j+8, N=lane-16
  const int rsel = (lane >> 4) * 8;
#pragma unroll
  for (int ni = 0; ni < 4; ++ni) {
    const int gcol = blockN + waveN * 64 + ni * 16 + (lane & 15);
    if (gcol >= N) continue;
    const float  sc = (float)S[gcol];
    const half_t bv = Bb[gcol];
#pragma unroll
    for (int mi = 0; mi < 2; ++mi) {
      const int rbase = blockM + waveM * 32 + mi * 16 + rsel;
#pragma unroll
      for (int j = 0; j < 8; ++j) {
        const int grow = rbase + j;
        if (grow < M)
          O[(uint64_t)grow * (uint32_t)N + (uint32_t)gcol] =
              (half_t)(acc[mi][ni][j] * sc) + bv;
      }
    }
  }
}

extern "C" void kernel_launch(void* const* d_in, const int* in_sizes, int n_in,
                              void* d_out, int out_size, void* d_ws, size_t ws_size,
                              hipStream_t stream) {
  // inputs (setup_inputs order): x[M,K] f16, weight[N,K] f16, scales[N] f16, bias[N] f16
  const half_t* X  = (const half_t*)d_in[0];
  const half_t* W  = (const half_t*)d_in[1];
  const half_t* S  = (const half_t*)d_in[2];
  const half_t* Bb = (const half_t*)d_in[3];
  half_t* O = (half_t*)d_out;   // fp16 output

  const int N = in_sizes[2];
  const int K = (int)((long long)in_sizes[1] / N);
  const int M = (int)((long long)in_sizes[0] / K);

  dim3 grid((N + BN - 1) / BN, (M + BM - 1) / BM);   // 86 x 64 for ref shapes
  fp6llm_gemm<<<grid, THREADS, 0, stream>>>(X, W, S, Bb, O, M, N, K);
}
